// Self_Channel_Attention_90460601188888
// MI455X (gfx1250) — matrix operand order repaired and checked by reading, not compiled, not run on hardware
//
#include <hip/hip_runtime.h>

#define BB 8
#define CC 256
#define NN 16384
#define KSPLIT 8

typedef __attribute__((ext_vector_type(16))) __bf16 bf16x16;
typedef __attribute__((ext_vector_type(8)))  __bf16 bf16x8;
typedef __attribute__((ext_vector_type(8)))  float  f32x8;

// ---------------- WMMA helpers (D layout per CDNA5 ISA 7.12.2; A/B use contiguous k = 16h + s) ----------------

__device__ __forceinline__ f32x8 wmma_bf16(bf16x16 a, bf16x16 b, f32x8 c) {
    return __builtin_amdgcn_wmma_f32_16x16x32_bf16(
        false, a, false, b, (short)0, c, false, false);
}

// A-matrix 16x32 bf16 tile from row-major src (rows = M, contiguous K).
// lane<16:  row M=lane,    K in {0..15}
// lane>=16: row M=lane-16, K in {16..31}
__device__ __forceinline__ bf16x16 load_A(const __bf16* base, int ld,
                                          int row0, int k0, int lane) {
    int r = lane & 15, h = lane >> 4;
    const __bf16* p = base + (size_t)(row0 + r) * ld + k0 + h * 16;
    bf16x8 lo = *(const bf16x8*)(p);
    bf16x8 hi = *(const bf16x8*)(p + 8);
    bf16x16 a;
#pragma unroll
    for (int t = 0; t < 8; ++t) { a[t] = lo[t]; a[t + 8] = hi[t]; }
    return a;
}

// B-matrix 32x16 bf16 tile where B[k, j'] = src[col0+j', k0+k] (src holds B^T,
// row-major, contiguous K). lane<16: col j'=lane, K=0..15; lane>=16: K=16..31.
__device__ __forceinline__ bf16x16 load_BT(const __bf16* base, int ld,
                                           int col0, int k0, int lane) {
    int r = lane & 15, h = lane >> 4;
    const __bf16* p = base + (size_t)(col0 + r) * ld + k0 + h * 16;
    return *(const bf16x16*)p;
}

// D 16x16 f32: lane covers col = lane%16, rows = (lane/16)*8 + e
__device__ __forceinline__ void store_D(float* base, int ld, int row0, int col0,
                                        f32x8 acc, int lane) {
    int r = lane & 15, h = lane >> 4;
#pragma unroll
    for (int e = 0; e < 8; ++e)
        base[(size_t)(row0 + h * 8 + e) * ld + col0 + r] = acc[e];
}

// ---------------- Stage 1: convert + transpose + row sums ----------------

// x [B,C,N] f32 -> xbf [B,C,N] bf16 and xbfT [B,N,C] bf16 (LDS 32x32 transpose)
__global__ __launch_bounds__(256) void kTransCvt(const float* __restrict__ x,
                                                 __bf16* __restrict__ xbf,
                                                 __bf16* __restrict__ xT) {
    __shared__ __bf16 tile[32][33];
    int b = blockIdx.z;
    int c0 = blockIdx.x * 32, n0 = blockIdx.y * 32;
    int tx = threadIdx.x, ty = threadIdx.y;  // 32 x 8
    const float* xb = x + ((size_t)b * CC + c0) * NN + n0;
    __bf16* xbfb = xbf + ((size_t)b * CC + c0) * NN + n0;
#pragma unroll
    for (int k = 0; k < 4; ++k) {
        int c = ty + k * 8;
        float v = xb[(size_t)c * NN + tx];
        __bf16 hv = (__bf16)v;
        xbfb[(size_t)c * NN + tx] = hv;
        tile[c][tx] = hv;
    }
    __syncthreads();
    __bf16* xTb = xT + ((size_t)b * NN + n0) * CC + c0;
#pragma unroll
    for (int k = 0; k < 4; ++k) {
        int n = ty + k * 8;
        xTb[(size_t)n * CC + tx] = tile[tx][n];
    }
}

// s[b,c] = sum_n x[b,c,n]
__global__ __launch_bounds__(256) void kRowSum(const float* __restrict__ x,
                                               float* __restrict__ s) {
    int row = blockIdx.x;  // b*C + c
    const float* xr = x + (size_t)row * NN;
    float acc = 0.f;
    for (int t = threadIdx.x; t < NN; t += 256) acc += xr[t];
    __shared__ float sm[256];
    sm[threadIdx.x] = acc;
    __syncthreads();
    for (int st = 128; st > 0; st >>= 1) {
        if (threadIdx.x < st) sm[threadIdx.x] += sm[threadIdx.x + st];
        __syncthreads();
    }
    if (threadIdx.x == 0) s[row] = sm[0];
}

// u[b,i] = Wk[i,:].s[b,:] ; w[b,i] = Wq[i,:].s[b,:]
__global__ __launch_bounds__(256) void kUW(const float* __restrict__ Wk,
                                           const float* __restrict__ Wq,
                                           const float* __restrict__ s,
                                           float* __restrict__ u,
                                           float* __restrict__ w) {
    int b = blockIdx.x, i = threadIdx.x;
    const float* sb = s + b * CC;
    float au = 0.f, aw = 0.f;
    for (int c = 0; c < CC; ++c) {
        float sv = sb[c];
        au += Wk[i * CC + c] * sv;
        aw += Wq[i * CC + c] * sv;
    }
    u[b * CC + i] = au;
    w[b * CC + i] = aw;
}

// ---------------- Stage 2: G = x x^T (bf16 WMMA, K split 8 ways) -----------
// One wave computes a 32x32 block of a partial Gram over a 2048-wide K slice.
// 512 jobs/batch * 8 batches = 4096 waves = 512 blocks. Deterministic
// (no float atomics): partials reduced by kReduceG.
__global__ __launch_bounds__(256) void kGram(const __bf16* __restrict__ xbf,
                                             float* __restrict__ Gpart) {
    int wave = threadIdx.x >> 5, lane = threadIdx.x & 31;
    int gw = blockIdx.x * 8 + wave;          // 4096 waves
    int b = gw >> 9;                         // 512 jobs per batch
    int rem = gw & 511;
    int ks = rem >> 6;                       // K-slice 0..7
    int rem2 = rem & 63;
    int i0 = (rem2 >> 3) * 32, j0 = (rem2 & 7) * 32;
    int nBeg = ks * (NN / KSPLIT), nEnd = nBeg + NN / KSPLIT;
    const __bf16* xb = xbf + (size_t)b * CC * NN;
    int r = lane & 15;
    f32x8 a00 = {}, a01 = {}, a10 = {}, a11 = {};
    for (int n0 = nBeg; n0 < nEnd; n0 += 32) {
        // warm L0/L1 one ~512B step ahead (gfx1250 global_prefetch_b8);
        // gate to every other iter (>=128B lines)
        if (((n0 >> 5) & 1) == 0 && n0 + 256 < nEnd) {
            __builtin_prefetch(xb + (size_t)(i0 + r) * NN + n0 + 256, 0, 1);
            __builtin_prefetch(xb + (size_t)(i0 + 16 + r) * NN + n0 + 256, 0, 1);
            __builtin_prefetch(xb + (size_t)(j0 + r) * NN + n0 + 256, 0, 1);
            __builtin_prefetch(xb + (size_t)(j0 + 16 + r) * NN + n0 + 256, 0, 1);
        }
        bf16x16 A0 = load_A(xb, NN, i0, n0, lane);
        bf16x16 A1 = load_A(xb, NN, i0 + 16, n0, lane);
        bf16x16 B0 = load_BT(xb, NN, j0, n0, lane);
        bf16x16 B1 = load_BT(xb, NN, j0 + 16, n0, lane);
        a00 = wmma_bf16(A0, B0, a00);
        a01 = wmma_bf16(A0, B1, a01);
        a10 = wmma_bf16(A1, B0, a10);
        a11 = wmma_bf16(A1, B1, a11);
    }
    float* Gp = Gpart + ((size_t)ks * BB + b) * CC * CC;
    store_D(Gp, CC, i0, j0, a00, lane);
    store_D(Gp, CC, i0, j0 + 16, a01, lane);
    store_D(Gp, CC, i0 + 16, j0, a10, lane);
    store_D(Gp, CC, i0 + 16, j0 + 16, a11, lane);
}

// G[idx] = sum_p Gpart[p][idx]  (deterministic reduction of K-split partials)
__global__ __launch_bounds__(256) void kReduceG(const float* __restrict__ Gpart,
                                                float* __restrict__ G) {
    size_t idx = (size_t)blockIdx.x * 256 + threadIdx.x;
    const size_t stride = (size_t)BB * CC * CC;
    float acc = 0.f;
#pragma unroll
    for (int p = 0; p < KSPLIT; ++p) acc += Gpart[p * stride + idx];
    G[idx] = acc;
}

// ---------------- Stage 3: small fp32 algebra (256^3, negligible) ----------

// T1[b,i,j] = sum_k Wk[i,k] * G[b,k,j]
__global__ __launch_bounds__(256) void kT1(const float* __restrict__ Wk,
                                           const float* __restrict__ G,
                                           float* __restrict__ T1) {
    int bi = blockIdx.x;
    int b = bi >> 8, i = bi & 255, j = threadIdx.x;
    const float* Gb = G + (size_t)b * CC * CC;
    float acc = 0.f;
    for (int k = 0; k < CC; ++k) acc += Wk[i * CC + k] * Gb[k * CC + j];
    T1[(size_t)bi * CC + j] = acc;
}

// energy[b,i,j] = sum_k T1[b,i,k]*Wq[j,k] + u[b,i]*bq[j] + bk[i]*w[b,j]
//                 + N*bk[i]*bq[j]
__global__ __launch_bounds__(256) void kEnergy(const float* __restrict__ T1,
                                               const float* __restrict__ Wq,
                                               const float* __restrict__ bq,
                                               const float* __restrict__ bk,
                                               const float* __restrict__ u,
                                               const float* __restrict__ w,
                                               float* __restrict__ energy) {
    int bi = blockIdx.x;
    int b = bi >> 8, i = bi & 255, j = threadIdx.x;
    const float* t1 = T1 + (size_t)bi * CC;
    float acc = 0.f;
    for (int k = 0; k < CC; ++k) acc += t1[k] * Wq[j * CC + k];
    acc += u[b * CC + i] * bq[j] + bk[i] * w[b * CC + j] +
           (float)NN * bk[i] * bq[j];
    energy[(size_t)bi * CC + j] = acc;
}

// in-place softmax over last axis (256 wide)
__global__ __launch_bounds__(256) void kSoftmax(float* __restrict__ energy) {
    int bi = blockIdx.x, j = threadIdx.x;
    float* row = energy + (size_t)bi * CC;
    __shared__ float sm[256];
    float v = row[j];
    sm[j] = v;
    __syncthreads();
    for (int st = 128; st > 0; st >>= 1) {
        if (j < st) sm[j] = fmaxf(sm[j], sm[j + st]);
        __syncthreads();
    }
    float m = sm[0];
    __syncthreads();
    float e = __expf(v - m);
    sm[j] = e;
    __syncthreads();
    for (int st = 128; st > 0; st >>= 1) {
        if (j < st) sm[j] += sm[j + st];
        __syncthreads();
    }
    row[j] = e / sm[0];
}

// Mbf[b,j,c] = bf16( sum_i att[b,i,j] * Wv[i,c] );  rvec[b,j] = sum_i att*bv
__global__ __launch_bounds__(256) void kMproj(const float* __restrict__ att,
                                              const float* __restrict__ Wv,
                                              const float* __restrict__ bv,
                                              __bf16* __restrict__ Mbf,
                                              float* __restrict__ rvec) {
    int bj = blockIdx.x;
    int b = bj >> 8, j = bj & 255, c = threadIdx.x;
    const float* attb = att + (size_t)b * CC * CC;
    float acc = 0.f, rv = 0.f;
    for (int i = 0; i < CC; ++i) {
        float a = attb[i * CC + j];  // uniform across block -> scalar load
        acc += a * Wv[i * CC + c];
        rv += a * bv[i];
    }
    Mbf[(size_t)bj * CC + c] = (__bf16)acc;
    if (c == 0) rvec[bj] = rv;
}

// ---------------- Stage 4: out^T = x^T M^T, fused residual ----------------
// One wave computes a 32(n) x 32(j) block of out^T[b]; epilogue writes
// final[b,j,n] = gamma*(acc + rvec[b,j]) + x[b,j,n] with contiguous stores.
__global__ __launch_bounds__(256) void kOut(const __bf16* __restrict__ xT,
                                            const __bf16* __restrict__ Mbf,
                                            const float* __restrict__ rvec,
                                            const float* __restrict__ x,
                                            const float* __restrict__ gamma,
                                            float* __restrict__ out) {
    int wave = threadIdx.x >> 5, lane = threadIdx.x & 31;
    int gw = blockIdx.x * 8 + wave;          // 32768 waves
    int b = gw >> 12;                        // 4096 waves per batch
    int rem = gw & 4095;
    int n0 = (rem >> 3) * 32;                // 512 n-blocks
    int j0 = (rem & 7) * 32;                 // 8 j-blocks
    const __bf16* xTb = xT + (size_t)b * NN * CC;
    const __bf16* Mb = Mbf + (size_t)b * CC * CC;
    f32x8 a00 = {}, a01 = {}, a10 = {}, a11 = {};
#pragma unroll
    for (int c0 = 0; c0 < CC; c0 += 32) {
        bf16x16 A0 = load_A(xTb, CC, n0, c0, lane);       // rows = n
        bf16x16 A1 = load_A(xTb, CC, n0 + 16, c0, lane);
        bf16x16 B0 = load_BT(Mb, CC, j0, c0, lane);       // B^T rows = M rows
        bf16x16 B1 = load_BT(Mb, CC, j0 + 16, c0, lane);
        a00 = wmma_bf16(A0, B0, a00);
        a01 = wmma_bf16(A0, B1, a01);
        a10 = wmma_bf16(A1, B0, a10);
        a11 = wmma_bf16(A1, B1, a11);
    }
    float g = gamma[0];
    int r = lane & 15, h = lane >> 4;
    f32x8 accs[2][2] = {{a00, a01}, {a10, a11}};
#pragma unroll
    for (int tn = 0; tn < 2; ++tn)
#pragma unroll
        for (int tj = 0; tj < 2; ++tj) {
            int j = j0 + tj * 16 + r;            // D col -> j
            int nb = n0 + tn * 16 + h * 8;       // D rows -> n (contiguous)
            size_t off = ((size_t)b * CC + j) * NN + nb;
            float rv = rvec[b * CC + j];
            f32x8 acc = accs[tn][tj];
#pragma unroll
            for (int e = 0; e < 8; ++e)
                out[off + e] = g * (acc[e] + rv) + x[off + e];
        }
}

// ---------------- launch ----------------

extern "C" void kernel_launch(void* const* d_in, const int* in_sizes, int n_in,
                              void* d_out, int out_size, void* d_ws,
                              size_t ws_size, hipStream_t stream) {
    const float* x     = (const float*)d_in[0];
    const float* Wq    = (const float*)d_in[1];
    const float* bq    = (const float*)d_in[2];
    const float* Wk    = (const float*)d_in[3];
    const float* bk    = (const float*)d_in[4];
    const float* Wv    = (const float*)d_in[5];
    const float* bv    = (const float*)d_in[6];
    const float* gamma = (const float*)d_in[7];
    float* out = (float*)d_out;

    char* ws = (char*)d_ws;
    size_t off = 0;
    __bf16* xbf = (__bf16*)(ws + off); off += (size_t)BB * CC * NN * 2;  // 64MB
    __bf16* xT  = (__bf16*)(ws + off); off += (size_t)BB * NN * CC * 2;  // 64MB
    float*  Gp  = (float*)(ws + off);  off += (size_t)KSPLIT * BB * CC * CC * 4;  // 16MB
    float*  G   = (float*)(ws + off);  off += (size_t)BB * CC * CC * 4;  // 2MB
    float*  T1  = (float*)(ws + off);  off += (size_t)BB * CC * CC * 4;  // 2MB
    float*  en  = (float*)(ws + off);  off += (size_t)BB * CC * CC * 4;  // 2MB
    __bf16* Mbf = (__bf16*)(ws + off); off += (size_t)BB * CC * CC * 2;  // 1MB
    float*  s   = (float*)(ws + off);  off += (size_t)BB * CC * 4;
    float*  u   = (float*)(ws + off);  off += (size_t)BB * CC * 4;
    float*  w   = (float*)(ws + off);  off += (size_t)BB * CC * 4;
    float*  rv  = (float*)(ws + off);  off += (size_t)BB * CC * 4;

    kTransCvt<<<dim3(CC / 32, NN / 32, BB), dim3(32, 8), 0, stream>>>(x, xbf, xT);
    kRowSum<<<BB * CC, 256, 0, stream>>>(x, s);
    kUW<<<BB, 256, 0, stream>>>(Wk, Wq, s, u, w);
    kGram<<<(KSPLIT * BB * 64) / 8, 256, 0, stream>>>(xbf, Gp);
    kReduceG<<<(BB * CC * CC) / 256, 256, 0, stream>>>(Gp, G);
    kT1<<<BB * CC, 256, 0, stream>>>(Wk, G, T1);
    kEnergy<<<BB * CC, 256, 0, stream>>>(T1, Wq, bq, bk, u, w, en);
    kSoftmax<<<BB * CC, 256, 0, stream>>>(en);
    kMproj<<<BB * CC, 256, 0, stream>>>(en, Wv, bv, Mbf, rv);
    kOut<<<(BB * 4096) / 8, 256, 0, stream>>>(xT, Mbf, rv, x, gamma, out);
}
